// Mamba3Block_36653250904126
// MI455X (gfx1250) — matrix operand order repaired and checked by reading, not compiled, not run on hardware
//
#include <hip/hip_runtime.h>
#include <hip/hip_bf16.h>
#include <math.h>

// ---------------------------------------------------------------------------
// Types for WMMA operands (CDNA5 / gfx1250, wave32)
// ---------------------------------------------------------------------------
typedef __attribute__((ext_vector_type(16))) __bf16          v16bf;
typedef __attribute__((ext_vector_type(8)))  float           v8f;
typedef __attribute__((ext_vector_type(8)))  unsigned short  ushort8;

__device__ __forceinline__ unsigned short f32_to_bf16_rne(float f) {
  unsigned u = __float_as_uint(f);
  u += 0x7FFFu + ((u >> 16) & 1u);      // round-to-nearest-even
  return (unsigned short)(u >> 16);
}

__device__ __forceinline__ v16bf pack_bf16x16(ushort8 lo, ushort8 hi) {
  union { ushort8 s[2]; v16bf v; } u;
  u.s[0] = lo; u.s[1] = hi;
  return u.v;
}

// LDS byte offset (relative to wave's LDS base) of a generic pointer into LDS
__device__ __forceinline__ unsigned lds_off_of(const void* p) {
  return (unsigned)(unsigned long long)
         (const __attribute__((address_space(3))) char*)p;
}

// CDNA5 async global->LDS copy (per-lane b32), tracked with ASYNCcnt.
__device__ __forceinline__ void async_load_f32_to_lds(unsigned lds_off,
                                                      const float* gptr) {
  asm volatile("global_load_async_to_lds_b32 %0, %1, off"
               :: "v"(lds_off), "v"((unsigned long long)gptr)
               : "memory");
}
__device__ __forceinline__ void wait_asynccnt_le1() {
  asm volatile("s_wait_asynccnt 0x1" ::: "memory");
}
__device__ __forceinline__ void wait_asynccnt_0() {
  asm volatile("s_wait_asynccnt 0x0" ::: "memory");
}

// ---------------------------------------------------------------------------
// Weight convert: f32 W[K,N] -> bf16 Wt[Npad,Kpad] (transposed, zero padded)
// ---------------------------------------------------------------------------
__global__ __launch_bounds__(256) void wcvt_t_kernel(
    const float* __restrict__ W, unsigned short* __restrict__ Wt,
    int K, int N, int Kpad, int Npad)
{
  int idx = blockIdx.x * 256 + threadIdx.x;
  int total = Kpad * Npad;
  if (idx >= total) return;
  int n = idx / Kpad;
  int k = idx - n * Kpad;
  float v = (n < N && k < K) ? W[(size_t)k * N + n] : 0.f;
  Wt[idx] = f32_to_bf16_rne(v);
}

// ---------------------------------------------------------------------------
// WMMA GEMM: C[M,N](f32) = A[M,K](bf16 row-major) x Bt[N,K](bf16)
// One wave computes a 16x64 strip: 4 accumulators share one A fragment
// (4x A reuse) and operands are software double-buffered so loads for
// k-step i+1 issue before the WMMAs of step i (hides L2 latency).
// Operand fetch matches the ISA VGPR layouts:
//   A 16x32: lanes 0-15 -> K=k0+{0..7,16..23}; lanes 16-31 -> K=k0+{8..15,24..31}
//   B 32x16: lanes 0-15 -> K=k0+{0..7,16..23}; lanes 16-31 -> K=k0+{8..15,24..31}
//   C/D f32: vgpr r -> M = 16*mt + r + (lane<16 ? 0 : 8), N = 16*nt + (lane&15)
// ---------------------------------------------------------------------------
__global__ __launch_bounds__(256) void gemm_bf16_wmma(
    const unsigned short* __restrict__ A,
    const unsigned short* __restrict__ Bt,
    float* __restrict__ C,
    int M, int N, int K, int lda, int ldb, int ldc)
{
  const int wave = (int)((blockIdx.x * blockDim.x + threadIdx.x) >> 5);
  const int lane = (int)(threadIdx.x & 31);
  const int mtiles  = M >> 4;
  const int ntiles  = N >> 4;
  const int ngroups = (ntiles + 3) >> 2;
  if (wave >= mtiles * ngroups) return;
  const int mt = wave / ngroups;
  const int ng = wave - mt * ngroups;
  const int l16 = lane & 15;
  const int hiL = lane >> 4;     // 0 for lanes 0-15, 1 for lanes 16-31

  const unsigned short* pa =
      A + (size_t)((mt << 4) + l16) * lda + (hiL ? 8 : 0);
  const unsigned short* pb[4];
  int ntj[4];
#pragma unroll
  for (int j = 0; j < 4; ++j) {
    int nt = (ng << 2) + j;
    if (nt > ntiles - 1) nt = ntiles - 1;   // clamp tail (redundant, benign)
    ntj[j] = nt;
    pb[j] = Bt + (size_t)((nt << 4) + l16) * ldb + (hiL ? 8 : 0);
  }

  const v8f z8 = {0.f, 0.f, 0.f, 0.f, 0.f, 0.f, 0.f, 0.f};
  v8f acc[4] = {z8, z8, z8, z8};

  // prologue: load k-step 0
  ushort8 a0 = *(const ushort8*)(pa);
  ushort8 a1 = *(const ushort8*)(pa + 16);
  ushort8 b0[4], b1[4];
#pragma unroll
  for (int j = 0; j < 4; ++j) {
    b0[j] = *(const ushort8*)(pb[j]);
    b1[j] = *(const ushort8*)(pb[j] + 16);
  }

  int k0 = 0;
  while (true) {
    const int  k1   = k0 + 32;
    const bool more = (k1 < K);

    // issue next-step loads before consuming current registers
    ushort8 na0 = a0, na1 = a1, nb0[4], nb1[4];
#pragma unroll
    for (int j = 0; j < 4; ++j) { nb0[j] = b0[j]; nb1[j] = b1[j]; }
    if (more) {
      na0 = *(const ushort8*)(pa + k1);
      na1 = *(const ushort8*)(pa + k1 + 16);
#pragma unroll
      for (int j = 0; j < 4; ++j) {
        nb0[j] = *(const ushort8*)(pb[j] + k1);
        nb1[j] = *(const ushort8*)(pb[j] + k1 + 16);
      }
    }
    // L2 prefetch two k-steps ahead (speculative; OOB dropped silently)
    __builtin_prefetch((const void*)(pa + k0 + 64), 0, 3);
    __builtin_prefetch((const void*)(pb[0] + k0 + 64), 0, 3);
    __builtin_prefetch((const void*)(pb[2] + k0 + 64), 0, 3);

    const v16bf av = pack_bf16x16(a0, a1);
#pragma unroll
    for (int j = 0; j < 4; ++j) {
      acc[j] = __builtin_amdgcn_wmma_f32_16x16x32_bf16(
          /*neg_a=*/false, av, /*neg_b=*/false, pack_bf16x16(b0[j], b1[j]),
          /*c_mod=*/(short)0, acc[j], /*reuse_a=*/false, /*reuse_b=*/false);
    }
    if (!more) break;
    a0 = na0; a1 = na1;
#pragma unroll
    for (int j = 0; j < 4; ++j) { b0[j] = nb0[j]; b1[j] = nb1[j]; }
    k0 = k1;
  }

#pragma unroll
  for (int j = 0; j < 4; ++j) {
    if ((ng << 2) + j < ntiles) {
      float* pc = C + (size_t)((mt << 4) + (hiL ? 8 : 0)) * ldc
                    + (ntj[j] << 4) + l16;
      const v8f a = acc[j];
#pragma unroll
      for (int r = 0; r < 8; ++r) pc[(size_t)r * ldc] = a[r];
    }
  }
}

// ---------------------------------------------------------------------------
// Block-wide f32 sum reduction (blockDim.x == 256, wave32)
// ---------------------------------------------------------------------------
__device__ __forceinline__ float block_reduce_sum_256(float v) {
  __shared__ float red[8];
  for (int o = 16; o > 0; o >>= 1) v += __shfl_down(v, o);
  if ((threadIdx.x & 31) == 0) red[threadIdx.x >> 5] = v;
  __syncthreads();
  if (threadIdx.x == 0) {
    float t = 0.f;
#pragma unroll
    for (int i = 0; i < 8; ++i) t += red[i];
    red[0] = t;
  }
  __syncthreads();
  return red[0];
}

// ---------------------------------------------------------------------------
// RMSNorm (+mask) producing bf16 activations. One block per row of D=512.
// ---------------------------------------------------------------------------
__global__ __launch_bounds__(256) void rmsnorm_mask_bf16_kernel(
    const float* __restrict__ x, const float* __restrict__ w,
    const float* __restrict__ mask, unsigned short* __restrict__ out, int D)
{
  const int row = blockIdx.x;
  const float* xr = x + (size_t)row * D;
  float s = 0.f;
  for (int i = threadIdx.x; i < D; i += 256) { float v = xr[i]; s += v * v; }
  float tot = block_reduce_sum_256(s);
  float sc = rsqrtf(tot / (float)D + 1e-6f) * mask[row];
  for (int i = threadIdx.x; i < D; i += 256)
    out[(size_t)row * D + i] = f32_to_bf16_rne(xr[i] * sc * w[i]);
}

// ---------------------------------------------------------------------------
// Fused residual-add + RMSNorm2: x1 = x + yout*mask ; xn2 = rmsnorm(x1)*w
// ---------------------------------------------------------------------------
__global__ __launch_bounds__(256) void resid_norm2_kernel(
    const float* __restrict__ x, const float* __restrict__ yout,
    const float* __restrict__ mask, const float* __restrict__ w,
    float* __restrict__ x1, unsigned short* __restrict__ xn2, int D)
{
  __shared__ float rowbuf[512];
  const int row = blockIdx.x;
  const float m = mask[row];
  float s = 0.f;
  for (int i = threadIdx.x; i < D; i += 256) {
    float v = x[(size_t)row * D + i] + yout[(size_t)row * D + i] * m;
    rowbuf[i] = v;
    x1[(size_t)row * D + i] = v;
    s += v * v;
  }
  __syncthreads();
  float tot = block_reduce_sum_256(s);
  float sc = rsqrtf(tot / (float)D + 1e-6f);
  for (int i = threadIdx.x; i < D; i += 256)
    xn2[(size_t)row * D + i] = f32_to_bf16_rne(rowbuf[i] * sc * w[i]);
}

// ---------------------------------------------------------------------------
// Selective-scan. One block per (b,h); 256 threads = 64 p-lanes x 4 rn-groups.
// State h[r,n,p] (128 x 64 f32) lives in registers (32 per thread).
// B/C tiles for step s+1 are brought into a double-buffered LDS stage with
// CDNA5 async global->LDS copies (ASYNCcnt) issued during step s's compute.
// Epilogue fuses  y = (C.h + u*D_skip) * silu(z)  -> bf16 for the Wout GEMM.
// ---------------------------------------------------------------------------
__global__ __launch_bounds__(256) void ssm_scan_kernel(
    const float* __restrict__ u,     // [BS, 1024] = [BS,H,P]
    const float* __restrict__ z,     // [BS, 1024]
    const float* __restrict__ Bm,    // [BS, 2048] = [BS,H,R,N]
    const float* __restrict__ Cm,    // [BS, 2048]
    const float* __restrict__ dtr,   // [BS, 16]
    const float* __restrict__ dt_bias,  // [16]
    const float* __restrict__ A_log,    // [16]
    const float* __restrict__ D_skip,   // [16]
    unsigned short* __restrict__ ygate, // [BS, 1024] bf16
    int S)
{
  const int b = blockIdx.x >> 4;   // / H
  const int h = blockIdx.x & 15;
  const int t = (int)threadIdx.x;
  const int p = t & 63;
  const int g = t >> 6;            // 0..3

  __shared__ float sB[2][128];
  __shared__ float sC[2][128];
  __shared__ float yred[256];

  // per-thread async staging role: waves 0-3 fetch B, waves 4-7 fetch C
  const int      isB  = (t < 128) ? 1 : 0;
  const int      e    = isB ? t : (t - 128);            // 0..127
  const float*   gsrc = isB ? Bm : Cm;
  const unsigned ldsA = lds_off_of(isB ? (const void*)&sB[0][e]
                                       : (const void*)&sC[0][e]);
  const unsigned ldsB = lds_off_of(isB ? (const void*)&sB[1][e]
                                       : (const void*)&sC[1][e]);

  float hs[32];
#pragma unroll
  for (int j = 0; j < 32; ++j) hs[j] = 0.f;

  const float negA = -expf(A_log[h]);
  const float bias = dt_bias[h];
  const float dsk  = D_skip[h];

  // prologue: stage step 0 into buffer 0
  async_load_f32_to_lds(ldsA, gsrc + (size_t)(b * S) * 2048 + h * 128 + e);

  for (int s = 0; s < S; ++s) {
    const int row = b * S + s;
    const int cur = s & 1;

    if (s + 1 < S) {
      // prefetch step s+1 into the other buffer, then wait for step s's data
      async_load_f32_to_lds(cur ? ldsA : ldsB,
                            gsrc + (size_t)(row + 1) * 2048 + h * 128 + e);
      wait_asynccnt_le1();
    } else {
      wait_asynccnt_0();
    }

    float draw = dtr[(size_t)row * 16 + h] + bias;
    float dt   = (draw > 20.f) ? draw : log1pf(expf(draw));   // softplus
    float a    = expf(negA * dt);                              // decay
    float uv   = u[(size_t)row * 1024 + h * 64 + p];
    float ud   = uv * dt;                                      // discretized input
    __syncthreads();   // step-s B/C visible to all waves; yred safe to rewrite

    float yp = 0.f;
#pragma unroll
    for (int j = 0; j < 32; ++j) {
      const int rn = g + (j << 2);
      hs[j] = a * hs[j] + sB[cur][rn] * ud;
      yp += sC[cur][rn] * hs[j];
    }
    yred[(g << 6) + p] = yp;
    __syncthreads();   // yred complete; all sB/sC reads of step s done

    if (g == 0) {
      float y = yred[p] + yred[64 + p] + yred[128 + p] + yred[192 + p];
      y += uv * dsk;
      float zv = z[(size_t)row * 1024 + h * 64 + p];
      float gate = zv / (1.f + expf(-zv));                    // silu(z)
      ygate[(size_t)row * 1024 + h * 64 + p] = f32_to_bf16_rne(y * gate);
    }
  }
}

// ---------------------------------------------------------------------------
// SwiGLU elementwise: h = silu(g1) * g3  (over padded [BS, 1376]) -> bf16
// ---------------------------------------------------------------------------
__global__ __launch_bounds__(256) void swiglu_kernel(
    const float* __restrict__ g1, const float* __restrict__ g3,
    unsigned short* __restrict__ hbuf, int total)
{
  int idx = blockIdx.x * 256 + threadIdx.x;
  if (idx >= total) return;
  float a = g1[idx];
  float v = (a / (1.f + expf(-a))) * g3[idx];
  hbuf[idx] = f32_to_bf16_rne(v);
}

// ---------------------------------------------------------------------------
// Final: out = (x1 + ffn) * mask
// ---------------------------------------------------------------------------
__global__ __launch_bounds__(256) void final_kernel(
    const float* __restrict__ x1, const float* __restrict__ ffn,
    const float* __restrict__ mask, float* __restrict__ out, int D, int total)
{
  int idx = blockIdx.x * 256 + threadIdx.x;
  if (idx >= total) return;
  int row = idx / D;
  out[idx] = (x1[idx] + ffn[idx]) * mask[row];
}

// ---------------------------------------------------------------------------
// Launcher
// ---------------------------------------------------------------------------
extern "C" void kernel_launch(void* const* d_in, const int* in_sizes, int n_in,
                              void* d_out, int out_size, void* d_ws, size_t ws_size,
                              hipStream_t stream) {
  (void)in_sizes; (void)n_in; (void)out_size; (void)ws_size;
  const float* x     = (const float*)d_in[0];
  const float* mask  = (const float*)d_in[1];
  const float* n1w   = (const float*)d_in[2];
  const float* n2w   = (const float*)d_in[3];
  const float* Wz    = (const float*)d_in[4];
  const float* Wx    = (const float*)d_in[5];
  const float* Wb    = (const float*)d_in[6];
  const float* Wc    = (const float*)d_in[7];
  const float* Wdt   = (const float*)d_in[8];
  const float* dtb   = (const float*)d_in[9];
  const float* Alog  = (const float*)d_in[10];
  const float* Dskip = (const float*)d_in[11];
  const float* Wout  = (const float*)d_in[12];
  const float* w1    = (const float*)d_in[13];
  const float* w2    = (const float*)d_in[14];
  const float* w3    = (const float*)d_in[15];
  float* out = (float*)d_out;

  const int BS = 2048;                 // B*S = 4*512  (S % CHUNK == 0, no pad)
  const int Dm = 512, DIN = 1024, NBC = 2048, DFF = 1368, DFFP = 1376, Hh = 16;
  const int S = 512;

  char* wsb = (char*)d_ws;
  size_t off = 0;
  auto alloc = [&](size_t bytes) -> void* {
    void* p = wsb + off;
    off += (bytes + 255) & ~(size_t)255;
    return p;
  };

  // bf16 transposed weights
  unsigned short* wz_t   = (unsigned short*)alloc((size_t)DIN  * Dm   * 2);
  unsigned short* wx_t   = (unsigned short*)alloc((size_t)DIN  * Dm   * 2);
  unsigned short* wb_t   = (unsigned short*)alloc((size_t)NBC  * Dm   * 2);
  unsigned short* wc_t   = (unsigned short*)alloc((size_t)NBC  * Dm   * 2);
  unsigned short* wdt_t  = (unsigned short*)alloc((size_t)Hh   * Dm   * 2);
  unsigned short* wout_t = (unsigned short*)alloc((size_t)Dm   * DIN  * 2);
  unsigned short* w1_t   = (unsigned short*)alloc((size_t)DFFP * Dm   * 2);
  unsigned short* w3_t   = (unsigned short*)alloc((size_t)DFFP * Dm   * 2);
  unsigned short* w2_t   = (unsigned short*)alloc((size_t)Dm   * DFFP * 2);
  // activations
  unsigned short* xn    = (unsigned short*)alloc((size_t)BS * Dm  * 2);
  float*          zb    = (float*)alloc((size_t)BS * DIN * 4);
  float*          ub    = (float*)alloc((size_t)BS * DIN * 4);
  float*          bm    = (float*)alloc((size_t)BS * NBC * 4);
  float*          cm    = (float*)alloc((size_t)BS * NBC * 4);
  float*          dtrw  = (float*)alloc((size_t)BS * Hh  * 4);
  unsigned short* ygate = (unsigned short*)alloc((size_t)BS * DIN * 2);
  float*          x1    = (float*)alloc((size_t)BS * Dm  * 4);
  unsigned short* xn2   = (unsigned short*)alloc((size_t)BS * Dm  * 2);
  unsigned short* hbuf  = (unsigned short*)alloc((size_t)BS * DFFP * 2);
  // phase-ordered aliases (buffers dead by the time they are reused)
  float* yout = zb;   // [BS, Dm]   <= z dead after scan
  float* g1   = bm;   // [BS, DFFP] <= Bm dead after scan
  float* g3   = cm;   // [BS, DFFP] <= Cm dead after scan
  float* ffn  = ub;   // [BS, Dm]   <= u dead after scan

  auto cvt = [&](const float* W, unsigned short* Wt, int K, int N, int Kp, int Np) {
    int total = Kp * Np;
    wcvt_t_kernel<<<(total + 255) / 256, 256, 0, stream>>>(W, Wt, K, N, Kp, Np);
  };
  cvt(Wz,   wz_t,   Dm,  DIN, Dm,   DIN);
  cvt(Wx,   wx_t,   Dm,  DIN, Dm,   DIN);
  cvt(Wb,   wb_t,   Dm,  NBC, Dm,   NBC);
  cvt(Wc,   wc_t,   Dm,  NBC, Dm,   NBC);
  cvt(Wdt,  wdt_t,  Dm,  Hh,  Dm,   Hh);
  cvt(Wout, wout_t, DIN, Dm,  DIN,  Dm);
  cvt(w1,   w1_t,   Dm,  DFF, Dm,   DFFP);
  cvt(w3,   w3_t,   Dm,  DFF, Dm,   DFFP);
  cvt(w2,   w2_t,   DFF, Dm,  DFFP, Dm);

  auto gemm = [&](const unsigned short* A, const unsigned short* Bt, float* C,
                  int M, int N, int K, int lda, int ldb, int ldc) {
    int ntiles  = N >> 4;
    int ngroups = (ntiles + 3) >> 2;
    int tiles   = (M >> 4) * ngroups;
    int blocks  = (tiles * 32 + 255) / 256;
    gemm_bf16_wmma<<<blocks, 256, 0, stream>>>(A, Bt, C, M, N, K, lda, ldb, ldc);
  };

  // 1) norm1 + mask -> bf16 xn
  rmsnorm_mask_bf16_kernel<<<BS, 256, 0, stream>>>(x, n1w, mask, xn, Dm);

  // 2) projections (all WMMA)
  gemm(xn, wz_t,  zb,   BS, DIN, Dm, Dm, Dm, DIN);
  gemm(xn, wx_t,  ub,   BS, DIN, Dm, Dm, Dm, DIN);
  gemm(xn, wb_t,  bm,   BS, NBC, Dm, Dm, Dm, NBC);
  gemm(xn, wc_t,  cm,   BS, NBC, Dm, Dm, Dm, NBC);
  gemm(xn, wdt_t, dtrw, BS, Hh,  Dm, Dm, Dm, Hh);

  // 3) sequential selective scan + D-skip + silu(z) gate -> bf16
  ssm_scan_kernel<<<64, 256, 0, stream>>>(ub, zb, bm, cm, dtrw,
                                          dtb, Alog, Dskip, ygate, S);

  // 4) output projection (WMMA)  [aliases zb]
  gemm(ygate, wout_t, yout, BS, Dm, DIN, DIN, DIN, Dm);

  // 5) residual + norm2
  resid_norm2_kernel<<<BS, 256, 0, stream>>>(x, yout, mask, n2w, x1, xn2, Dm);

  // 6) FFN up projections (WMMA, padded N=1376) [aliases bm/cm]
  gemm(xn2, w1_t, g1, BS, DFFP, Dm, Dm, Dm, DFFP);
  gemm(xn2, w3_t, g3, BS, DFFP, Dm, Dm, Dm, DFFP);

  // 7) SwiGLU -> bf16 (zero-padded cols stay zero: silu(0)*0 == 0)
  {
    int total = BS * DFFP;
    swiglu_kernel<<<(total + 255) / 256, 256, 0, stream>>>(g1, g3, hbuf, total);
  }

  // 8) FFN down projection (WMMA, K padded to 1376) [aliases ub]
  gemm(hbuf, w2_t, ffn, BS, Dm, DFFP, DFFP, DFFP, Dm);

  // 9) final residual + mask
  {
    int total = BS * Dm;
    final_kernel<<<(total + 255) / 256, 256, 0, stream>>>(x1, ffn, mask, out, Dm, total);
  }
}
